// Scalble_DAT_Net_57604101374251
// MI455X (gfx1250) — matrix operand order repaired and checked by reading, not compiled, not run on hardware
//
#include <hip/hip_runtime.h>

typedef __attribute__((ext_vector_type(16))) _Float16 v16h;
typedef __attribute__((ext_vector_type(8)))  float    v8f;

#define N_NODES 100000
#define N_EDGES 1000000
#define HDIM    64
#define NGRAPH  256
#define NCLASS  8
#define ALPHA   0.5f

// ---------------- degree / normalization ----------------
__global__ void k_fill1(float* __restrict__ p, int n) {
    int i = blockIdx.x * blockDim.x + threadIdx.x;
    if (i < n) p[i] = 1.0f;   // self-loop contribution to degree
}

__global__ void k_deg(const int* __restrict__ row, float* __restrict__ deg, int e) {
    int i = blockIdx.x * blockDim.x + threadIdx.x;
    if (i < e) atomicAdd(&deg[row[i]], 1.0f);
}

__global__ void k_dinv(float* __restrict__ d, int n) {
    int i = blockIdx.x * blockDim.x + threadIdx.x;
    if (i < n) d[i] = rsqrtf(fmaxf(d[i], 1.0f));
}

// ---------------- SGC linear: h0 = x @ W + b via WMMA f16 ----------------
// W is pre-swizzled into LDS in the exact WMMA B-operand register layout so
// the inner loop issues one contiguous 32B LDS load per lane (2x ds_load_b128)
// instead of 16x ds_load_u16.
__global__ __launch_bounds__(128)
void k_sgc_gemm(const float* __restrict__ x, const float* __restrict__ W,
                const float* __restrict__ b, float* __restrict__ h0, int n)
{
    // Ws_sw[((c*4+tt)*32 + lane)*16 + j] = W[c*32 + (lane>>4)*8 + (j&7) + (j>>3)*16][tt*16 + (lane&15)]
    __shared__ _Float16 Ws_sw[64 * 64];   // 8 KB
    int t = threadIdx.x;
    #pragma unroll
    for (int i = 0; i < 32; ++i) {
        int idx  = t + i * 128;
        int j    = idx & 15;
        int lane = (idx >> 4) & 31;
        int ctt  = idx >> 9;              // 0..7
        int c    = ctt >> 2;
        int tt   = ctt & 3;
        int krow = c * 32 + ((lane >> 4) << 3) + (j & 7) + ((j >> 3) << 4);
        int ncol = tt * 16 + (lane & 15);
        Ws_sw[idx] = (_Float16)W[krow * 64 + ncol];
    }
    __syncthreads();

    int wave = t >> 5, lane = t & 31;
    int row0 = (blockIdx.x * 4 + wave) * 16;
    if (row0 >= n) return;                    // wave-uniform: EXEC all-ones inside

    int m  = lane & 15;                       // output row (A) / col (B,C,D) index
    int hi = lane >> 4;

    const float* xr = x + (size_t)(row0 + m) * 64;

    v8f acc[4] = {};
    #pragma unroll
    for (int c = 0; c < 2; ++c) {             // two K=32 chunks cover K=64
        // A 16x32 f16 layout: lane half 'hi' holds K = kb..kb+7 and kb+16..kb+23
        int kb = c * 32 + hi * 8;
        const float4 p0 = *reinterpret_cast<const float4*>(xr + kb);
        const float4 p1 = *reinterpret_cast<const float4*>(xr + kb + 4);
        const float4 p2 = *reinterpret_cast<const float4*>(xr + kb + 16);
        const float4 p3 = *reinterpret_cast<const float4*>(xr + kb + 20);
        v16h a;
        a[0]  = (_Float16)p0.x; a[1]  = (_Float16)p0.y;
        a[2]  = (_Float16)p0.z; a[3]  = (_Float16)p0.w;
        a[4]  = (_Float16)p1.x; a[5]  = (_Float16)p1.y;
        a[6]  = (_Float16)p1.z; a[7]  = (_Float16)p1.w;
        a[8]  = (_Float16)p2.x; a[9]  = (_Float16)p2.y;
        a[10] = (_Float16)p2.z; a[11] = (_Float16)p2.w;
        a[12] = (_Float16)p3.x; a[13] = (_Float16)p3.y;
        a[14] = (_Float16)p3.z; a[15] = (_Float16)p3.w;

        #pragma unroll
        for (int tt = 0; tt < 4; ++tt) {      // 4 N-tiles of 16 -> H=64
            const v16h bm = *reinterpret_cast<const v16h*>(
                &Ws_sw[(((c * 4 + tt) * 32 + lane) << 4)]);
            acc[tt] = __builtin_amdgcn_wmma_f32_16x16x32_f16(
                false, a, false, bm, (short)0, acc[tt], false, false);
        }
    }

    #pragma unroll
    for (int tt = 0; tt < 4; ++tt) {
        int ncol   = tt * 16 + m;
        float bias = b[ncol];
        #pragma unroll
        for (int r = 0; r < 8; ++r) {
            int mr = r + hi * 8;              // C/D layout: lanes 0-15 -> M=r, 16-31 -> M=8+r
            h0[(size_t)(row0 + mr) * 64 + ncol] = acc[tt][r] + bias;
        }
    }
}

// ---------------- propagation hop: scatter + combine ----------------
__global__ void k_scatter(const int* __restrict__ row, const int* __restrict__ col,
                          const float* __restrict__ dinv, const float* __restrict__ hp,
                          float* __restrict__ agg, int e)
{
    int idx = blockIdx.x * blockDim.x + threadIdx.x;   // e * 16 threads
    int ei  = idx >> 4;
    if (ei >= e) return;
    int fc = (idx & 15) * 4;
    int r = row[ei], c = col[ei];
    float nm = dinv[r] * dinv[c];
    const float4 hv = *reinterpret_cast<const float4*>(hp + (size_t)c * 64 + fc);
    float* ap = agg + (size_t)r * 64 + fc;
    atomicAdd(ap + 0, nm * hv.x);
    atomicAdd(ap + 1, nm * hv.y);
    atomicAdd(ap + 2, nm * hv.z);
    atomicAdd(ap + 3, nm * hv.w);
}

__global__ void k_combine(float* __restrict__ dst, const float* __restrict__ hp,
                          const float* __restrict__ h0, const float* __restrict__ dinv,
                          int n, int do_relu)
{
    int i = blockIdx.x * blockDim.x + threadIdx.x;     // n * 64 elements
    if (i >= n * 64) return;
    int node = i >> 6;
    float di = dinv[node];
    // self-loop term dinv^2 * h_prev folded in here (not scattered)
    float v = (1.0f - ALPHA) * (dst[i] + di * di * hp[i]) + ALPHA * h0[i];
    if (do_relu) v = fmaxf(v, 0.0f);
    dst[i] = v;
}

// ---------------- readout pass 1: v, vmax, gmp ----------------
__global__ void k_att1(const float* __restrict__ h, const float* __restrict__ watt,
                       const float* __restrict__ batt, const int* __restrict__ batch,
                       float* __restrict__ v, float* __restrict__ vmax,
                       float* __restrict__ gmp, int n)
{
    int node = blockIdx.x * 8 + (threadIdx.x >> 5);
    int lane = threadIdx.x & 31;
    if (node >= n) return;                  // wave-uniform exit
    const float* hr = h + (size_t)node * 64;
    float h0v = hr[lane], h1v = hr[lane + 32];
    float p = h0v * watt[lane] + h1v * watt[lane + 32];
    #pragma unroll
    for (int off = 16; off > 0; off >>= 1)
        p += __shfl_xor(p, off, 32);
    int g = batch[node];
    if (lane == 0) {
        float vv = 1.0f / (1.0f + expf(-(p + batt[0])));
        v[node] = vv;
        atomicMax(reinterpret_cast<unsigned int*>(&vmax[g]), __float_as_uint(vv));
    }
    // h >= 0 after relu, so uint atomicMax == float max
    atomicMax(reinterpret_cast<unsigned int*>(&gmp[g * 64 + lane]),      __float_as_uint(h0v));
    atomicMax(reinterpret_cast<unsigned int*>(&gmp[g * 64 + lane + 32]), __float_as_uint(h1v));
}

// ---------------- readout pass 2: ve, vsum ----------------
__global__ void k_att2(float* __restrict__ v, const float* __restrict__ vmax,
                       float* __restrict__ vsum, const int* __restrict__ batch, int n)
{
    int i = blockIdx.x * blockDim.x + threadIdx.x;
    if (i >= n) return;
    int g = batch[i];
    float ve = expf(v[i] - vmax[g]);
    v[i] = ve;
    atomicAdd(&vsum[g], ve);
}

// ---------------- readout pass 3: gsp ----------------
__global__ void k_att3(const float* __restrict__ h, const float* __restrict__ v,
                       const float* __restrict__ vsum, const int* __restrict__ batch,
                       float* __restrict__ gsp, int n)
{
    int node = blockIdx.x * 8 + (threadIdx.x >> 5);
    int lane = threadIdx.x & 31;
    if (node >= n) return;
    int g = batch[node];
    float w = v[node] / (vsum[g] + 1e-16f);
    const float* hr = h + (size_t)node * 64;
    atomicAdd(&gsp[g * 64 + lane],      w * hr[lane]);
    atomicAdd(&gsp[g * 64 + lane + 32], w * hr[lane + 32]);
}

// ---------------- MLP head + log_softmax ----------------
__global__ __launch_bounds__(128)
void k_head(const float* __restrict__ gmp, const float* __restrict__ gsp,
            const float* __restrict__ W1, const float* __restrict__ b1,
            const float* __restrict__ W2, const float* __restrict__ b2,
            const float* __restrict__ W3, const float* __restrict__ b3,
            float* __restrict__ out)
{
    __shared__ float z[128], z1[64], z2[32], z3[8];
    int g = blockIdx.x, t = threadIdx.x;
    z[t] = (t < 64) ? gmp[g * 64 + t] : gsp[g * 64 + (t - 64)];
    __syncthreads();
    if (t < 64) {
        float s = b1[t];
        for (int k = 0; k < 128; ++k) s += z[k] * W1[k * 64 + t];
        z1[t] = fmaxf(s, 0.0f);
    }
    __syncthreads();
    if (t < 32) {
        float s = b2[t];
        for (int k = 0; k < 64; ++k) s += z1[k] * W2[k * 32 + t];
        z2[t] = fmaxf(s, 0.0f);
    }
    __syncthreads();
    if (t < 8) {
        float s = b3[t];
        for (int k = 0; k < 32; ++k) s += z2[k] * W3[k * 8 + t];
        z3[t] = s;
    }
    __syncthreads();
    if (t == 0) {
        float mx = z3[0];
        for (int j = 1; j < 8; ++j) mx = fmaxf(mx, z3[j]);
        float sum = 0.0f;
        for (int j = 0; j < 8; ++j) sum += expf(z3[j] - mx);
        float lse = mx + logf(sum);
        for (int j = 0; j < 8; ++j) out[g * 8 + j] = z3[j] - lse;
    }
}

extern "C" void kernel_launch(void* const* d_in, const int* in_sizes, int n_in,
                              void* d_out, int out_size, void* d_ws, size_t ws_size,
                              hipStream_t stream)
{
    (void)in_sizes; (void)n_in; (void)out_size; (void)ws_size;

    const float* x     = (const float*)d_in[0];
    const int*   ei    = (const int*)  d_in[1];
    const int*   batch = (const int*)  d_in[2];
    const float* W_sgc = (const float*)d_in[3];
    const float* b_sgc = (const float*)d_in[4];
    const float* w_att = (const float*)d_in[5];
    const float* b_att = (const float*)d_in[6];
    const float* W1    = (const float*)d_in[7];
    const float* b1    = (const float*)d_in[8];
    const float* W2    = (const float*)d_in[9];
    const float* b2    = (const float*)d_in[10];
    const float* W3    = (const float*)d_in[11];
    const float* b3    = (const float*)d_in[12];
    const int* row = ei;
    const int* col = ei + N_EDGES;
    float* out = (float*)d_out;

    // ---- workspace carve-out (256B aligned) ----
    char* ws = (char*)d_ws;
    size_t off = 0;
    auto carve = [&](size_t bytes) -> float* {
        float* p = (float*)(ws + off);
        off = (off + bytes + 255) & ~(size_t)255;
        return p;
    };
    const size_t NH = (size_t)N_NODES * HDIM * sizeof(float);
    float* h0   = carve(NH);
    float* hA   = carve(NH);
    float* hB   = carve(NH);
    float* dinv = carve((size_t)N_NODES * sizeof(float));
    float* v    = carve((size_t)N_NODES * sizeof(float));
    float* vmax = carve((size_t)NGRAPH * sizeof(float));
    float* vsum = carve((size_t)NGRAPH * sizeof(float));
    float* gmp  = carve((size_t)NGRAPH * HDIM * sizeof(float));
    float* gsp  = carve((size_t)NGRAPH * HDIM * sizeof(float));

    // ---- degrees / normalization ----
    k_fill1<<<(N_NODES + 255) / 256, 256, 0, stream>>>(dinv, N_NODES);
    k_deg  <<<(N_EDGES + 255) / 256, 256, 0, stream>>>(row, dinv, N_EDGES);
    k_dinv <<<(N_NODES + 255) / 256, 256, 0, stream>>>(dinv, N_NODES);

    // ---- SGC linear (WMMA) ----
    k_sgc_gemm<<<(N_NODES / 16 + 3) / 4, 128, 0, stream>>>(x, W_sgc, b_sgc, h0, N_NODES);

    // ---- K=3 propagation hops, ping-pong hA/hB ----
    const float* prev[3] = { h0, hA, hB };
    float*       dst [3] = { hA, hB, hA };
    for (int k = 0; k < 3; ++k) {
        hipMemsetAsync(dst[k], 0, NH, stream);
        k_scatter<<<(N_EDGES * 16) / 256, 256, 0, stream>>>(row, col, dinv, prev[k], dst[k], N_EDGES);
        k_combine<<<(N_NODES * HDIM) / 256, 256, 0, stream>>>(dst[k], prev[k], h0, dinv, N_NODES, k == 2);
    }
    float* h = hA;  // final relu(h)

    // ---- readout ----
    hipMemsetAsync(vmax, 0, (size_t)NGRAPH * sizeof(float), stream);
    hipMemsetAsync(vsum, 0, (size_t)NGRAPH * sizeof(float), stream);
    hipMemsetAsync(gmp,  0, (size_t)NGRAPH * HDIM * sizeof(float), stream);
    hipMemsetAsync(gsp,  0, (size_t)NGRAPH * HDIM * sizeof(float), stream);

    k_att1<<<(N_NODES + 7) / 8, 256, 0, stream>>>(h, w_att, b_att, batch, v, vmax, gmp, N_NODES);
    k_att2<<<(N_NODES + 255) / 256, 256, 0, stream>>>(v, vmax, vsum, batch, N_NODES);
    k_att3<<<(N_NODES + 7) / 8, 256, 0, stream>>>(h, v, vsum, batch, gsp, N_NODES);

    // ---- MLP head ----
    k_head<<<NGRAPH, 128, 0, stream>>>(gmp, gsp, W1, b1, W2, b2, W3, b3, out);
}
